// MultiHeadAttention_60352880444004
// MI455X (gfx1250) — matrix operand order repaired and checked by reading, not compiled, not run on hardware
//
#include <hip/hip_runtime.h>

typedef __attribute__((ext_vector_type(4)))  _Float16 v4h;
typedef __attribute__((ext_vector_type(8)))  _Float16 v8h;
typedef __attribute__((ext_vector_type(16))) _Float16 v16h;
typedef __attribute__((ext_vector_type(8)))  float    v8f;

#define BATCH 4
#define SEQ   2048
#define DIM   1024
#define NHEAD 16
#define HD    64
#define BT    (BATCH*SEQ)

// ---------------------------------------------------------------------------
// CDNA5 async copy: global -> LDS, 16B per lane, tracked by ASYNCcnt.
// LDS destination address comes from the dst VGPR; generic LDS pointer low 32 bits = LDS offset.
// ---------------------------------------------------------------------------
static __device__ __forceinline__ unsigned lds_off(const void* p) {
  return (unsigned)(size_t)p;
}
static __device__ __forceinline__ void async_cp16(unsigned lds, const void* g) {
  asm volatile("global_load_async_to_lds_b128 %0, %1, off"
               :: "v"(lds), "v"((unsigned long long)(size_t)g) : "memory");
}
#define WAIT_ASYNC(n) asm volatile("s_wait_asynccnt " #n ::: "memory")
#define WAIT_DS0()    asm volatile("s_wait_dscnt 0" ::: "memory")

// ---------------------------------------------------------------------------
// WMMA fragment loaders (16x16x32 f16; A/B share contiguous k = 16h + s order, C/D per ISA):
//  A (16x32): lane L<16 -> row L, K=k0..k0+15; hi half shifted +16
//  B (32x16): lane n<16 -> col n, K=k0..k0+15 (one contiguous 32B row)
//  C/D (16x16 f32): VGPR r -> row r (lanes 0-15) / r+8 (lanes 16-31), lane%16=col
// ---------------------------------------------------------------------------
static __device__ __forceinline__ v16h load_a_frag(const _Float16* __restrict__ p,
                                                   int ld, int m0, int k0) {
  const int lane = threadIdx.x & 31;
  const _Float16* q = p + (m0 + (lane & 15)) * ld + k0 + ((lane >> 4) << 4);
  v8h lo = *(const v8h*)(q);
  v8h hi = *(const v8h*)(q + 8);
  v16h r;
#pragma unroll
  for (int i = 0; i < 8; ++i) { r[i] = lo[i]; r[i + 8] = hi[i]; }
  return r;
}

static __device__ __forceinline__ v16h load_b_frag(const _Float16* p, int ld,
                                                   int n0, int k0) {
  const int lane = threadIdx.x & 31;
  const _Float16* q = p + (n0 + (lane & 15)) * ld + k0 + ((lane >> 4) << 4);
  return *(const v16h*)(q);
}

static __device__ __forceinline__ v8f wmma_f16(v16h a, v16h b, v8f c) {
  return __builtin_amdgcn_wmma_f32_16x16x32_f16(false, a, false, b, (short)0, c,
                                                false, false);
}

// ---------------------------------------------------------------------------
// f32 -> f16 cast (4 elems / thread)
// ---------------------------------------------------------------------------
__global__ __launch_bounds__(256) void cast_f32_to_f16_kernel(
    const float* __restrict__ in, _Float16* __restrict__ out, int n4) {
  const int i = blockIdx.x * blockDim.x + threadIdx.x;
  if (i < n4) {
    const float4 v = ((const float4*)in)[i];
    v4h o;
    o[0] = (_Float16)v.x; o[1] = (_Float16)v.y;
    o[2] = (_Float16)v.z; o[3] = (_Float16)v.w;
    ((v4h*)out)[i] = o;
  }
}

// ---------------------------------------------------------------------------
// NT GEMM: out[m,n] = sum_k A[m,k]*W[n,k] + bias[n]
// Block = 8 waves sharing one 64-col W panel (staged in LDS via async copy,
// double buffered); wave w owns rows m0+16w .. m0+16w+15.
// MODE 0: f16 [B,H,T,Hd] (Q/K);  MODE 1: f16 [B,H,Hd,T] (V^T);
// MODE 2: f32 [BT,DIM] (output projection)
// ---------------------------------------------------------------------------
template <int MODE>
__global__ __launch_bounds__(256) void gemm_nt_kernel(
    const _Float16* __restrict__ A, const _Float16* __restrict__ W,
    const float* __restrict__ bias, _Float16* __restrict__ outh,
    float* __restrict__ outf) {
  __shared__ _Float16 wl[2][64 * 32];  // 2 x 4KB W tile [64 rows][32 k]
  const int tid  = threadIdx.x;
  const int lane = tid & 31;
  const int wid  = tid >> 5;
  const int m0 = (blockIdx.x >> 4) * 128 + wid * 16;  // 64 M-blocks of 128
  const int n0 = (blockIdx.x & 15) << 6;              // 16 N-blocks of 64

  // stage: 256 threads x 16B = 4KB; one async op per thread per stage
  auto stage = [&](int buf, int k0) {
    const int row = tid >> 2, part = tid & 3;
    async_cp16(lds_off(&wl[buf][0]) + (unsigned)tid * 16,
               W + (size_t)(n0 + row) * DIM + k0 + part * 8);
  };

  stage(0, 0);
  v8f acc[4] = {};
  for (int k0 = 0, it = 0; k0 < DIM; k0 += 32, ++it) {
    const int buf = it & 1;
    if (k0 + 32 < DIM) { stage(buf ^ 1, k0 + 32); WAIT_ASYNC(1); }
    else               { WAIT_ASYNC(0); }
    __syncthreads();
    v16h a = load_a_frag(A, DIM, m0, k0);
#pragma unroll
    for (int j = 0; j < 4; ++j) {
      v16h b = load_b_frag(&wl[buf][0], 32, j * 16, 0);
      acc[j] = wmma_f16(a, b, acc[j]);
    }
    __syncthreads();
  }

  const int rowoff = (lane >> 4) << 3;
  const int cl     = lane & 15;
#pragma unroll
  for (int j = 0; j < 4; ++j) {
    const int n  = n0 + j * 16 + cl;
    const float bv = bias[n];
    const int h = n >> 6;
    const int d = n & 63;
#pragma unroll
    for (int r = 0; r < 8; ++r) {
      const int row = m0 + r + rowoff;
      const float val = acc[j][r] + bv;
      if (MODE == 0) {
        const int bidx = row >> 11, t = row & (SEQ - 1);
        outh[(((size_t)(bidx * NHEAD + h) * SEQ + t) << 6) + d] = (_Float16)val;
      } else if (MODE == 1) {
        const int bidx = row >> 11, t = row & (SEQ - 1);
        outh[(((size_t)(bidx * NHEAD + h) * HD + d) << 11) + t] = (_Float16)val;
      } else {
        outf[(size_t)row * DIM + n] = val;
      }
    }
  }
}

// ---------------------------------------------------------------------------
// Flash attention. Block = 8 waves sharing one (b,h); each wave owns a
// 16-query tile. K/V tiles (64 keys) staged per block into LDS with async
// copies (double buffered). Online softmax over 64-key steps.
// Q,K: [B,H,T,Hd] f16;  Vt: [B,H,Hd,T] f16;  O: [B,T,D] f16.
// ---------------------------------------------------------------------------
__global__ __launch_bounds__(256) void flash_attn_kernel(
    const _Float16* __restrict__ Q, const _Float16* __restrict__ Km,
    const _Float16* __restrict__ Vt, _Float16* __restrict__ O) {
  __shared__ _Float16 kl[2][64 * 64];    // [key][d]   2 x 8KB
  __shared__ _Float16 vl[2][64 * 64];    // [d][key]   2 x 8KB
  __shared__ _Float16 plds[8][16 * 64];  // per-wave P tile, 16KB total
  const int tid  = threadIdx.x;
  const int lane = tid & 31;
  const int wid  = tid >> 5;
  const int h    = (blockIdx.x >> 4) & 15;   // 8 waves/block share (b,h)
  const int bidx = blockIdx.x >> 8;
  const int qt   = ((blockIdx.x * 8 + wid) & 127);

  const _Float16* Qh = Q  + (size_t)((bidx * NHEAD + h) * SEQ) * HD;
  const _Float16* Kh = Km + (size_t)((bidx * NHEAD + h) * SEQ) * HD;
  const _Float16* Vh = Vt + (size_t)((bidx * NHEAD + h) * HD) * SEQ;
  const int q0 = qt << 4;

  const v16h qa0 = load_a_frag(Qh, HD, q0, 0);
  const v16h qa1 = load_a_frag(Qh, HD, q0, 32);

  // stage 64-key K and V tiles: 2 x 8KB = 1024 chunks; 4 async ops / thread
  auto stage = [&](int buf, int kt) {
#pragma unroll
    for (int s = 0; s < 2; ++s) {
      const int c = tid + s * 256;
      const int row = c >> 3, part = c & 7;
      async_cp16(lds_off(&kl[buf][0]) + (unsigned)c * 16,
                 Kh + (size_t)(kt + row) * HD + part * 8);
      async_cp16(lds_off(&vl[buf][0]) + (unsigned)c * 16,
                 Vh + (size_t)row * SEQ + kt + part * 8);
    }
  };

  v8f o[4] = {};
  float mrow[8], lrow[8];
#pragma unroll
  for (int r = 0; r < 8; ++r) { mrow[r] = -1e30f; lrow[r] = 0.f; }

  _Float16* pl   = &plds[wid][0];
  const int cl     = lane & 15;
  const int rowoff = (lane >> 4) << 3;

  stage(0, 0);
  for (int kt = 0, it = 0; kt < SEQ; kt += 64, ++it) {
    const int buf = it & 1;
    if (kt + 64 < SEQ) { stage(buf ^ 1, kt + 64); WAIT_ASYNC(4); }
    else               { WAIT_ASYNC(0); }
    __syncthreads();

    // ---- S = Q*K^T for 64 keys: four 16x16 f32 tiles from LDS K ----
    v8f s[4];
#pragma unroll
    for (int t = 0; t < 4; ++t) {
      v8f sz = {};
      v16h b0 = load_b_frag(&kl[buf][0], 64, t * 16, 0);
      v16h b1 = load_b_frag(&kl[buf][0], 64, t * 16, 32);
      sz = wmma_f16(qa0, b0, sz);
      sz = wmma_f16(qa1, b1, sz);
      s[t] = sz;
    }

    // ---- online softmax; rows of a tile live across one 16-lane half ----
#pragma unroll
    for (int r = 0; r < 8; ++r) {
      const float a0 = s[0][r] * 0.125f, a1 = s[1][r] * 0.125f;
      const float a2 = s[2][r] * 0.125f, a3 = s[3][r] * 0.125f;
      float mx = fmaxf(fmaxf(a0, a1), fmaxf(a2, a3));
      mx = fmaxf(mx, __shfl_xor(mx, 1, 32));
      mx = fmaxf(mx, __shfl_xor(mx, 2, 32));
      mx = fmaxf(mx, __shfl_xor(mx, 4, 32));
      mx = fmaxf(mx, __shfl_xor(mx, 8, 32));
      const float mn  = fmaxf(mrow[r], mx);
      const float fac = __expf(mrow[r] - mn);
      mrow[r] = mn;
      const float p0 = __expf(a0 - mn), p1 = __expf(a1 - mn);
      const float p2 = __expf(a2 - mn), p3 = __expf(a3 - mn);
      float rs = (p0 + p1) + (p2 + p3);
      rs += __shfl_xor(rs, 1, 32);
      rs += __shfl_xor(rs, 2, 32);
      rs += __shfl_xor(rs, 4, 32);
      rs += __shfl_xor(rs, 8, 32);
      lrow[r] = lrow[r] * fac + rs;
      o[0][r] *= fac; o[1][r] *= fac; o[2][r] *= fac; o[3][r] *= fac;
      _Float16* pr = pl + (r + rowoff) * 64 + cl;
      pr[0] = (_Float16)p0; pr[16] = (_Float16)p1;
      pr[32] = (_Float16)p2; pr[48] = (_Float16)p3;
    }
    WAIT_DS0();
    // ---- reload P (16x64) as two A fragments ----
    v16h pa0, pa1;
    {
      const _Float16* q = pl + (lane & 15) * 64 + ((lane >> 4) << 4);
      v8h lo0 = *(const v8h*)(q);
      v8h hi0 = *(const v8h*)(q + 8);
      v8h lo1 = *(const v8h*)(q + 32);
      v8h hi1 = *(const v8h*)(q + 40);
#pragma unroll
      for (int i = 0; i < 8; ++i) {
        pa0[i] = lo0[i]; pa0[i + 8] = hi0[i];
        pa1[i] = lo1[i]; pa1[i + 8] = hi1[i];
      }
    }
    // ---- O += P * V from LDS V^T tile ----
#pragma unroll
    for (int j = 0; j < 4; ++j) {
      v16h v0 = load_b_frag(&vl[buf][0], 64, j * 16, 0);
      v16h v1 = load_b_frag(&vl[buf][0], 64, j * 16, 32);
      o[j] = wmma_f16(pa0, v0, o[j]);
      o[j] = wmma_f16(pa1, v1, o[j]);
    }
    __syncthreads();
  }

  float inv[8];
#pragma unroll
  for (int r = 0; r < 8; ++r) inv[r] = 1.0f / lrow[r];
#pragma unroll
  for (int j = 0; j < 4; ++j) {
#pragma unroll
    for (int r = 0; r < 8; ++r) {
      const int t = q0 + r + rowoff;
      O[(size_t)(bidx * SEQ + t) * DIM + h * HD + j * 16 + cl] =
          (_Float16)(o[j][r] * inv[r]);
    }
  }
}

// ---------------------------------------------------------------------------
extern "C" void kernel_launch(void* const* d_in, const int* in_sizes, int n_in,
                              void* d_out, int out_size, void* d_ws,
                              size_t ws_size, hipStream_t stream) {
  (void)in_sizes; (void)n_in; (void)out_size; (void)ws_size;
  const float* x  = (const float*)d_in[0];
  const float* Wq = (const float*)d_in[1];
  const float* bq = (const float*)d_in[2];
  const float* Wk = (const float*)d_in[3];
  const float* bk = (const float*)d_in[4];
  const float* Wv = (const float*)d_in[5];
  const float* bv = (const float*)d_in[6];
  const float* Wo = (const float*)d_in[7];
  const float* bo = (const float*)d_in[8];

  _Float16* ws  = (_Float16*)d_ws;
  _Float16* xh  = ws;                         // BT*DIM
  _Float16* wqh = xh  + (size_t)BT * DIM;     // DIM*DIM each
  _Float16* wkh = wqh + (size_t)DIM * DIM;
  _Float16* wvh = wkh + (size_t)DIM * DIM;
  _Float16* woh = wvh + (size_t)DIM * DIM;
  _Float16* Qs  = woh + (size_t)DIM * DIM;    // BT*DIM each
  _Float16* Ks  = Qs  + (size_t)BT * DIM;
  _Float16* Vts = Ks  + (size_t)BT * DIM;
  _Float16* At  = Vts + (size_t)BT * DIM;

  const int xq4 = BT * DIM / 4;
  const int wq4 = DIM * DIM / 4;
  cast_f32_to_f16_kernel<<<xq4 / 256, 256, 0, stream>>>(x, xh, xq4);
  cast_f32_to_f16_kernel<<<wq4 / 256, 256, 0, stream>>>(Wq, wqh, wq4);
  cast_f32_to_f16_kernel<<<wq4 / 256, 256, 0, stream>>>(Wk, wkh, wq4);
  cast_f32_to_f16_kernel<<<wq4 / 256, 256, 0, stream>>>(Wv, wvh, wq4);
  cast_f32_to_f16_kernel<<<wq4 / 256, 256, 0, stream>>>(Wo, woh, wq4);

  const int gemm_blocks = (BT / 128) * (DIM / 64);  // 1024
  gemm_nt_kernel<0><<<gemm_blocks, 256, 0, stream>>>(xh, wqh, bq, Qs, nullptr);
  gemm_nt_kernel<0><<<gemm_blocks, 256, 0, stream>>>(xh, wkh, bk, Ks, nullptr);
  gemm_nt_kernel<1><<<gemm_blocks, 256, 0, stream>>>(xh, wvh, bv, Vts, nullptr);

  const int attn_blocks = (BATCH * NHEAD * (SEQ / 16)) / 8;  // 1024
  flash_attn_kernel<<<attn_blocks, 256, 0, stream>>>(Qs, Ks, Vts, At);

  gemm_nt_kernel<2><<<gemm_blocks, 256, 0, stream>>>(At, woh, bo, nullptr,
                                                     (float*)d_out);
}
